// TAM_53214644798214
// MI455X (gfx1250) — compile-verified, matrix operand order checked by reading, not run on hardware
//
#include <hip/hip_runtime.h>
#include <hip/hip_bf16.h>

// Problem constants (match reference)
#define N_   256
#define C_   256
#define V_   25
#define T_   20
#define K_   3
#define O_   256
#define C4_  64
#define VT_  (V_ * T_)   // 500
#define EPSF 1e-5f

typedef __attribute__((ext_vector_type(16))) __bf16 v16bf;
typedef __attribute__((ext_vector_type(8)))  float  v8f;

union FragAB { v16bf v; unsigned u[8]; };

// ---------------------------------------------------------------------------
// Kernel 1: pooled[n,c,t] = mean over v of x[n,c,v,t]
// ---------------------------------------------------------------------------
__global__ __launch_bounds__(256) void pool_kernel(const float* __restrict__ x,
                                                   float* __restrict__ pooled) {
    int idx = blockIdx.x * blockDim.x + threadIdx.x;   // (n*C + c)*T + t
    if (idx >= N_ * C_ * T_) return;
    int t  = idx % T_;
    int nc = idx / T_;
    const float* p = x + (size_t)nc * VT_ + t;
    float s = 0.f;
    #pragma unroll
    for (int v = 0; v < V_; ++v) s += p[v * T_];
    pooled[idx] = s * (1.0f / V_);
}

// ---------------------------------------------------------------------------
// Kernel 2: G branch (dynamic kernel weights) + L branch (excitation), per n
// ---------------------------------------------------------------------------
__global__ __launch_bounds__(256) void branch_kernel(
    const float* __restrict__ pooled,  // (N,C,T)
    const float* __restrict__ Wg1,     // (40,20)
    const float* __restrict__ Wg2,     // (3,40)
    const float* __restrict__ Wl1,     // (64,256,3)
    const float* __restrict__ Wl2,     // (256,64)
    const float* __restrict__ bng,     // (4,40)
    const float* __restrict__ bnl,     // (4,64)
    float* __restrict__ kern_out,      // (N,C,3)
    float* __restrict__ l_out)         // (N,C,T)
{
    const int n   = blockIdx.x;
    const int tid = threadIdx.x;

    __shared__ float s_p [C_ ][21];    // pad 20->21: kills stride-20 bank conflicts
    __shared__ float s_l1[C4_][21];

    {
        const float* pr = pooled + ((size_t)n * C_ + tid) * T_;
        #pragma unroll
        for (int t = 0; t < T_; ++t) s_p[tid][t] = pr[t];
    }
    __syncthreads();

    // ---- G branch: per channel c = tid ----
    {
        const int c = tid;
        float h[40];
        #pragma unroll 4
        for (int j = 0; j < 40; ++j) {
            float a0 = 0.f;
            #pragma unroll
            for (int t = 0; t < T_; ++t) a0 += Wg1[j * T_ + t] * s_p[c][t];
            float g = bng[j], b = bng[40 + j], m = bng[80 + j], vv = bng[120 + j];
            float hv = (a0 - m) * (g * rsqrtf(vv + EPSF)) + b;
            h[j] = hv > 0.f ? hv : 0.f;
        }
        float sc[3];
        #pragma unroll
        for (int k = 0; k < 3; ++k) {
            float a0 = 0.f;
            #pragma unroll 8
            for (int j = 0; j < 40; ++j) a0 += Wg2[k * 40 + j] * h[j];
            sc[k] = a0;
        }
        float mx = fmaxf(sc[0], fmaxf(sc[1], sc[2]));
        float e0 = __expf(sc[0] - mx), e1 = __expf(sc[1] - mx), e2 = __expf(sc[2] - mx);
        float inv = 1.f / (e0 + e1 + e2);
        float* ko = kern_out + ((size_t)n * C_ + c) * 3;
        ko[0] = e0 * inv; ko[1] = e1 * inv; ko[2] = e2 * inv;
    }

    // ---- L branch conv1 (C -> C4, k=3, pad 1) + bn + relu ----
    {
        const int c4 = tid >> 2;
        const int t0 = (tid & 3) * 5;
        float a5[5] = {0.f, 0.f, 0.f, 0.f, 0.f};
        for (int c = 0; c < C_; ++c) {
            const float* w = Wl1 + ((size_t)c4 * C_ + c) * 3;
            float w0 = w[0], w1 = w[1], w2 = w[2];
            const float* pr = s_p[c];
            #pragma unroll
            for (int i = 0; i < 5; ++i) {
                int t = t0 + i;
                float pm = (t > 0)      ? pr[t - 1] : 0.f;
                float pp = (t < T_ - 1) ? pr[t + 1] : 0.f;
                a5[i] += w0 * pm + w1 * pr[t] + w2 * pp;
            }
        }
        float g = bnl[c4], b = bnl[64 + c4], m = bnl[128 + c4], vv = bnl[192 + c4];
        float s = g * rsqrtf(vv + EPSF);
        #pragma unroll
        for (int i = 0; i < 5; ++i) {
            float hv = (a5[i] - m) * s + b;
            s_l1[c4][t0 + i] = hv > 0.f ? hv : 0.f;
        }
    }
    __syncthreads();

    // ---- L branch conv2 (C4 -> C) + sigmoid ----
    {
        const int c = tid;
        float aT[T_];
        #pragma unroll
        for (int t = 0; t < T_; ++t) aT[t] = 0.f;
        for (int c4 = 0; c4 < C4_; ++c4) {
            float w = Wl2[(size_t)c * C4_ + c4];
            #pragma unroll
            for (int t = 0; t < T_; ++t) aT[t] += w * s_l1[c4][t];
        }
        float* lo = l_out + ((size_t)n * C_ + c) * T_;
        #pragma unroll
        for (int t = 0; t < T_; ++t) lo[t] = 1.f / (1.f + __expf(-aT[t]));
    }
}

// ---------------------------------------------------------------------------
// Kernel 3: split Wc2 into bf16 hi + lo (hi/lo split-precision GEMM operands)
// ---------------------------------------------------------------------------
__global__ __launch_bounds__(256) void cvt_split_kernel(const float* __restrict__ w,
                                                        __bf16* __restrict__ whi,
                                                        __bf16* __restrict__ wlo,
                                                        int nElem) {
    int i = blockIdx.x * blockDim.x + threadIdx.x;
    if (i >= nElem) return;
    float f  = w[i];
    __bf16 h = (__bf16)f;
    whi[i] = h;
    wlo[i] = (__bf16)(f - (float)h);
}

// ---------------------------------------------------------------------------
// Kernel 4: fused dynamic depthwise conv + bn1 + relu + (Wc2 GEMM) + bn2 + relu
// Grid: (8 col-tiles of 64) x (N).  8 waves/WG; wave w owns output rows
// [32w,32w+32) x 64 cols as 2x4 WMMA accumulator tiles.
// GEMM: out[o,col] = sum_c Wc2[o,c] * z[c,col],  col = v*T + t in [0,500)
// z computed on the fly, double-buffered into LDS as bf16 hi/lo; x taps for
// the next K-chunk are register-prefetched so global latency overlaps WMMA.
// ---------------------------------------------------------------------------
__global__ __launch_bounds__(256) void main_gemm_kernel(
    const float*  __restrict__ x,        // (N,C,V,T)
    const float*  __restrict__ l_in,     // (N,C,T)
    const float*  __restrict__ kern_in,  // (N,C,3)
    const __bf16* __restrict__ Whi,      // (O,C) bf16 hi
    const __bf16* __restrict__ Wlo,      // (O,C) bf16 lo
    const float*  __restrict__ bn1p,     // (4,C)
    const float*  __restrict__ bn2p,     // (4,O)
    float* __restrict__ out)             // (N,O,V,T) == (n*O+o)*500 + col
{
    const int tile = blockIdx.x;           // 0..7
    const int n    = blockIdx.y;           // 0..255
    const int tid  = threadIdx.x;
    const int lane = tid & 31;
    const int wave = tid >> 5;             // 0..7
    const int colbase = tile * 64;

    __shared__ float s_l[C_][T_];
    __shared__ float s_kern[C_][3];
    __shared__ float s_s1[C_], s_b1[C_], s_s2[O_], s_b2[O_];
    // stride 34 (gcd(17,64)=1) -> conflict-free fragment reads; double-buffered
    __shared__ __align__(16) __bf16 zhi[2][64 * 34];
    __shared__ __align__(16) __bf16 zlo[2][64 * 34];

    // ---- per-n staging (one channel per thread) ----
    {
        const float* lp = l_in + ((size_t)n * C_ + tid) * T_;
        #pragma unroll
        for (int t = 0; t < T_; ++t) s_l[tid][t] = lp[t];
        const float* kp = kern_in + ((size_t)n * C_ + tid) * 3;
        s_kern[tid][0] = kp[0]; s_kern[tid][1] = kp[1]; s_kern[tid][2] = kp[2];

        float g1 = bn1p[tid], b1 = bn1p[C_ + tid], m1 = bn1p[2 * C_ + tid], v1 = bn1p[3 * C_ + tid];
        float s1 = g1 * rsqrtf(v1 + EPSF);
        s_s1[tid] = s1;  s_b1[tid] = b1 - m1 * s1;

        float g2 = bn2p[tid], b2 = bn2p[O_ + tid], m2 = bn2p[2 * O_ + tid], v2 = bn2p[3 * O_ + tid];
        float s2 = g2 * rsqrtf(v2 + EPSF);
        s_s2[tid] = s2;  s_b2[tid] = b2 - m2 * s2;
    }
    __syncthreads();   // s_l / s_kern / bn scales ready before first stage()

    v8f acc[2][4];
    {
        v8f zero = {};
        #pragma unroll
        for (int i = 0; i < 2; ++i)
            #pragma unroll
            for (int j = 0; j < 4; ++j) acc[i][j] = zero;
    }

    const int c_local = tid >> 3;          // 0..31 (K rows staged per thread)
    const int colgrp  = tid & 7;           // 0..7  (adjacent threads -> adjacent cols)

    // CDNA5 wave32 16-bit A/B fragment k-map: pairs {2p,2p+1}+kb and {16+2p,..}+kb
    const int kb = (lane >> 4) * 8;        // lanes 16-31 hold k+8 halves
    int kmap[8];
    #pragma unroll
    for (int p = 0; p < 8; ++p) kmap[p] = (p < 4 ? 2 * p : 16 + 2 * (p - 4)) + kb;

    // --- register prefetch of x taps (3 per col) for one 32-channel chunk ---
    float xv[8][3], xnext[8][3];
    {
        const float* xrow = x + ((size_t)n * C_ + c_local) * VT_;   // c0 = 0
        #pragma unroll
        for (int j = 0; j < 8; ++j) {
            int col = colbase + j * 8 + colgrp;
            float a = 0.f, b = 0.f, d = 0.f;
            if (col < VT_) {
                int v = col / T_;
                int t = col - v * T_;
                const float* xr = xrow + v * T_;
                b = xr[t];
                if (t > 0)      a = xr[t - 1];
                if (t < T_ - 1) d = xr[t + 1];
            }
            xv[j][0] = a; xv[j][1] = b; xv[j][2] = d;
        }
    }

    int pb = 0;
    for (int c0 = 0; c0 < C_; c0 += 32) {
        // ---- stage z chunk from prefetched regs: VALU + ds_store only ----
        {
            int c = c0 + c_local;
            float k0 = s_kern[c][0], k1 = s_kern[c][1], k2 = s_kern[c][2];
            float s1 = s_s1[c], b1 = s_b1[c];
            const float* lrow = s_l[c];
            #pragma unroll
            for (int j = 0; j < 8; ++j) {
                int col_local = j * 8 + colgrp;
                int col = colbase + col_local;
                float z = 0.f;
                if (col < VT_) {
                    int v = col / T_;
                    int t = col - v * T_;
                    float pre = k1 * (xv[j][1] + lrow[t]);
                    if (t > 0)      pre += k0 * (xv[j][0] + lrow[t - 1]);
                    if (t < T_ - 1) pre += k2 * (xv[j][2] + lrow[t + 1]);
                    z = pre * s1 + b1;
                    z = z > 0.f ? z : 0.f;
                }
                __bf16 h = (__bf16)z;
                zhi[pb][col_local * 34 + c_local] = h;
                zlo[pb][col_local * 34 + c_local] = (__bf16)(z - (float)h);
            }
        }
        __syncthreads();   // buffer pb visible; also fences reads of pb^1 two iters back

        // ---- issue next chunk's x loads early (overlap with WMMA below) ----
        if (c0 + 32 < C_) {
            const float* xrow = x + ((size_t)n * C_ + (c0 + 32 + c_local)) * VT_;
            #pragma unroll
            for (int j = 0; j < 8; ++j) {
                int col = colbase + j * 8 + colgrp;
                float a = 0.f, b = 0.f, d = 0.f;
                if (col < VT_) {
                    int v = col / T_;
                    int t = col - v * T_;
                    const float* xr = xrow + v * T_;
                    b = xr[t];
                    if (t > 0)      a = xr[t - 1];
                    if (t < T_ - 1) d = xr[t + 1];
                }
                xnext[j][0] = a; xnext[j][1] = b; xnext[j][2] = d;
            }
        }

        // ---- build fragments ----
        FragAB ah[2], al[2];
        #pragma unroll
        for (int rt = 0; rt < 2; ++rt) {
            int m = wave * 32 + rt * 16 + (lane & 15);
            const __bf16* wh = Whi + (size_t)m * C_ + c0;
            const __bf16* wl = Wlo + (size_t)m * C_ + c0;
            #pragma unroll
            for (int p = 0; p < 8; ++p) {
                ah[rt].u[p] = *(const unsigned*)(wh + kmap[p]);
                al[rt].u[p] = *(const unsigned*)(wl + kmap[p]);
            }
        }
        FragAB bh[4], bl[4];
        #pragma unroll
        for (int ct = 0; ct < 4; ++ct) {
            int cbase = (ct * 16 + (lane & 15)) * 34;
            #pragma unroll
            for (int p = 0; p < 8; ++p) {
                bh[ct].u[p] = *(const unsigned*)(&zhi[pb][0] + cbase + kmap[p]);
                bl[ct].u[p] = *(const unsigned*)(&zlo[pb][0] + cbase + kmap[p]);
            }
        }

        // ---- 3-term split-precision WMMA: Wh*Bh + Wh*Bl + Wl*Bh ----
        #pragma unroll
        for (int rt = 0; rt < 2; ++rt)
            #pragma unroll
            for (int ct = 0; ct < 4; ++ct) {
                acc[rt][ct] = __builtin_amdgcn_wmma_f32_16x16x32_bf16(
                    false, ah[rt].v, false, bh[ct].v, (short)0, acc[rt][ct], false, false);
                acc[rt][ct] = __builtin_amdgcn_wmma_f32_16x16x32_bf16(
                    false, ah[rt].v, false, bl[ct].v, (short)0, acc[rt][ct], false, false);
                acc[rt][ct] = __builtin_amdgcn_wmma_f32_16x16x32_bf16(
                    false, al[rt].v, false, bh[ct].v, (short)0, acc[rt][ct], false, false);
            }

        // rotate prefetch registers / flip LDS buffer
        #pragma unroll
        for (int j = 0; j < 8; ++j) {
            xv[j][0] = xnext[j][0]; xv[j][1] = xnext[j][1]; xv[j][2] = xnext[j][2];
        }
        pb ^= 1;
    }

    // ---- epilogue: folded bn2 + relu, coalesced stores ----
    const int ncol_lane = lane & 15;
    const int mrow_off  = (lane >> 4) * 8;      // D layout: VGPR r -> M = r + 8*(lane>=16)
    #pragma unroll
    for (int rt = 0; rt < 2; ++rt) {
        #pragma unroll
        for (int ct = 0; ct < 4; ++ct) {
            int col = colbase + ct * 16 + ncol_lane;
            if (col >= VT_) continue;
            #pragma unroll
            for (int r = 0; r < 8; ++r) {
                int o = wave * 32 + rt * 16 + mrow_off + r;
                float val = acc[rt][ct][r] * s_s2[o] + s_b2[o];
                val = val > 0.f ? val : 0.f;
                out[((size_t)n * O_ + o) * VT_ + col] = val;
            }
        }
    }
}

// ---------------------------------------------------------------------------
// Host launcher
// ---------------------------------------------------------------------------
static inline size_t align256(size_t v) { return (v + 255) & ~(size_t)255; }

extern "C" void kernel_launch(void* const* d_in, const int* in_sizes, int n_in,
                              void* d_out, int out_size, void* d_ws, size_t ws_size,
                              hipStream_t stream) {
    (void)in_sizes; (void)n_in; (void)out_size; (void)ws_size;

    const float* x    = (const float*)d_in[0];
    const float* Wg1  = (const float*)d_in[1];
    const float* Wg2  = (const float*)d_in[2];
    const float* Wl1  = (const float*)d_in[3];
    const float* Wl2  = (const float*)d_in[4];
    const float* Wc2  = (const float*)d_in[5];
    const float* bng  = (const float*)d_in[6];
    const float* bnl  = (const float*)d_in[7];
    const float* bn1p = (const float*)d_in[8];
    const float* bn2p = (const float*)d_in[9];
    float* out = (float*)d_out;

    char* ws = (char*)d_ws;
    float*  pooled = (float*)ws;  ws += align256((size_t)N_ * C_ * T_ * sizeof(float));
    float*  kern   = (float*)ws;  ws += align256((size_t)N_ * C_ * 3  * sizeof(float));
    float*  l      = (float*)ws;  ws += align256((size_t)N_ * C_ * T_ * sizeof(float));
    __bf16* whi    = (__bf16*)ws; ws += align256((size_t)O_ * C_ * sizeof(__bf16));
    __bf16* wlo    = (__bf16*)ws; ws += align256((size_t)O_ * C_ * sizeof(__bf16));

    pool_kernel<<<(N_ * C_ * T_ + 255) / 256, 256, 0, stream>>>(x, pooled);
    branch_kernel<<<N_, 256, 0, stream>>>(pooled, Wg1, Wg2, Wl1, Wl2, bng, bnl, kern, l);
    cvt_split_kernel<<<(O_ * C_ + 255) / 256, 256, 0, stream>>>(Wc2, whi, wlo, O_ * C_);
    main_gemm_kernel<<<dim3(8, N_), 256, 0, stream>>>(x, l, kern, whi, wlo, bn1p, bn2p, out);
}
